// SaintEncoder_8065948582459
// MI455X (gfx1250) — matrix operand order repaired and checked by reading, not compiled, not run on hardware
//
#include <hip/hip_runtime.h>
#include <hip/hip_bf16.h>

// ---------------------------------------------------------------------------
// SAINT-style 4-layer encoder for MI455X (gfx1250), bf16 WMMA everywhere,
// async global->LDS DMA (ASYNCcnt) double-buffered GEMM pipeline.
// B=32 S=512 D=512 H=8 DH=64 F=2048 L=4, M = B*S = 16384 tokens.
// ---------------------------------------------------------------------------

#define B_ 32
#define S_ 512
#define D_ 512
#define H_ 8
#define DH_ 64
#define F_ 2048
#define L_ 4
#define M_ (B_ * S_)

typedef __attribute__((ext_vector_type(16))) __bf16 v16bf;
typedef __attribute__((ext_vector_type(8)))  float  v8f;

union Frag { v16bf v; uint4 q[2]; };
union U2B  { uint2 u; __bf16 h[4]; };

static __device__ inline v8f wmma_bf16(v16bf a, v16bf b, v8f c) {
  // D = A(16x32 bf16) * B(32x16 bf16) + C(16x16 f32)
  return __builtin_amdgcn_wmma_f32_16x16x32_bf16(false, a, false, b,
                                                 (short)0, c, false, false);
}

// CDNA5 async memory->LDS copy (no VGPR bounce, tracked by ASYNCcnt).
#define ASYNC_B128(ldsaddr, gaddr)                                   \
  asm volatile("global_load_async_to_lds_b128 %0, %1, off"           \
               :: "v"(ldsaddr), "v"(gaddr) : "memory")
#define WAIT_ASYNC(n)                                                \
  asm volatile("s_wait_asynccnt " #n ::: "memory")

static __device__ inline unsigned lds_addr(const void* p) {
  // AS(3) flat address: low 32 bits are the LDS byte offset.
  return (unsigned)(unsigned long long)p;
}

// ---------------------------------------------------------------------------
// Elementwise f32 -> bf16 conversion (4 elems / thread)
// ---------------------------------------------------------------------------
__global__ __launch_bounds__(256) void cvt_f32_bf16_kernel(
    const float* __restrict__ in, __bf16* __restrict__ out, long long n) {
  long long i = ((long long)blockIdx.x * 256 + threadIdx.x) * 4;
  if (i + 3 < n) {
#pragma unroll
    for (int k = 0; k < 4; ++k) out[i + k] = (__bf16)in[i + k];
  }
}

// ---------------------------------------------------------------------------
// W[K][N] f32  ->  Wt[N][K] bf16  (pre-transpose so GEMM B-frags are
// contiguous along K per column, matching WMMA B-matrix lane layout)
// ---------------------------------------------------------------------------
__global__ __launch_bounds__(256) void wtrans_kernel(
    const float* __restrict__ W, __bf16* __restrict__ Wt, int K, int N) {
  long long idx = (long long)blockIdx.x * 256 + threadIdx.x;
  if (idx < (long long)N * K) {
    int n = (int)(idx / K);
    int k = (int)(idx % K);
    Wt[idx] = (__bf16)W[(long long)k * N + n];
  }
}

// ---------------------------------------------------------------------------
// GEMM: C[M][N] = A[M][K](bf16) * Wt[N][K](bf16) + bias[N]  (+epilogue)
//   mode 0: out bf16 = acc + bias
//   mode 1: out f32  = acc + bias + residual   (pre-layernorm sum)
//   mode 2: out bf16 = gelu(acc + bias)        (exact erf gelu)
// Workgroup tile 128x128, 8 waves, each wave 32x64 (2x4 WMMA tiles).
// Ping/pong LDS buffers (K-tile 32), statically indexed, filled by async
// global->LDS b128 DMA; each thread owns 4 async ops per tile and retires
// the tile about to be consumed with s_wait_asynccnt 4 (in-order DMA).
// K is always a multiple of 64 here, so each loop iteration handles two
// K-tiles with fixed buffer roles (no runtime parity).
// ---------------------------------------------------------------------------
#define GTM 128
#define GTN 128
#define GTK 32
#define LSTR 40

__global__ __launch_bounds__(256) void gemm_bf16_kernel(
    const __bf16* __restrict__ A, const __bf16* __restrict__ Bt,
    const float* __restrict__ bias, const float* __restrict__ residual,
    void* __restrict__ outp, int M, int N, int K, int mode) {
  __shared__ __bf16 As[2][GTM * LSTR];
  __shared__ __bf16 Bs[2][GTN * LSTR];

  const int tid  = threadIdx.x;
  const int lane = tid & 31;
  const int wave = tid >> 5;
  const int half = lane >> 4;   // 0: lanes 0-15, 1: lanes 16-31
  const int l16  = lane & 15;
  const int wm   = wave & 3;    // 4 waves along M (each 32 rows)
  const int wn   = wave >> 2;   // 2 waves along N (each 64 cols)
  const int m0   = blockIdx.y * GTM;
  const int n0   = blockIdx.x * GTN;

  const int lc = tid & 3;       // 8-elem chunk within K-tile
  const int lr = tid >> 2;      // row 0..63 (two passes -> 128 rows)

  // Per-thread staging bases (row pointers reused every tile).
  const __bf16* gA0 = A  + (size_t)(m0 + lr) * K      + lc * 8;
  const __bf16* gA1 = A  + (size_t)(m0 + lr + 64) * K + lc * 8;
  const __bf16* gB0 = Bt + (size_t)(n0 + lr) * K      + lc * 8;
  const __bf16* gB1 = Bt + (size_t)(n0 + lr + 64) * K + lc * 8;
  const unsigned lA0[2] = {lds_addr(&As[0][lr * LSTR + lc * 8]),
                           lds_addr(&As[1][lr * LSTR + lc * 8])};
  const unsigned lA1[2] = {lds_addr(&As[0][(lr + 64) * LSTR + lc * 8]),
                           lds_addr(&As[1][(lr + 64) * LSTR + lc * 8])};
  const unsigned lB0[2] = {lds_addr(&Bs[0][lr * LSTR + lc * 8]),
                           lds_addr(&Bs[1][lr * LSTR + lc * 8])};
  const unsigned lB1[2] = {lds_addr(&Bs[0][(lr + 64) * LSTR + lc * 8]),
                           lds_addr(&Bs[1][(lr + 64) * LSTR + lc * 8])};

  v8f acc[2][4];
#pragma unroll
  for (int i = 0; i < 2; ++i)
#pragma unroll
    for (int j = 0; j < 4; ++j)
      acc[i][j] = (v8f){0.f, 0.f, 0.f, 0.f, 0.f, 0.f, 0.f, 0.f};

#define GEMM_ISSUE(bufi, kk)                                         \
  do {                                                               \
    ASYNC_B128(lA0[bufi], gA0 + (kk));                               \
    ASYNC_B128(lB0[bufi], gB0 + (kk));                               \
    ASYNC_B128(lA1[bufi], gA1 + (kk));                               \
    ASYNC_B128(lB1[bufi], gB1 + (kk));                               \
  } while (0)

#define GEMM_COMPUTE(bufi)                                           \
  do {                                                               \
    Frag a[2];                                                       \
    _Pragma("unroll") for (int i = 0; i < 2; ++i) {                  \
      int row = wm * 32 + i * 16 + l16;                              \
      a[i].q[0] = *(const uint4*)&As[bufi][row * LSTR + half * 8];   \
      a[i].q[1] = *(const uint4*)&As[bufi][row * LSTR + 16 + half * 8]; \
    }                                                                \
    Frag b[4];                                                       \
    _Pragma("unroll") for (int j = 0; j < 4; ++j) {                  \
      int row = wn * 64 + j * 16 + l16;                              \
      b[j].q[0] = *(const uint4*)&Bs[bufi][row * LSTR + half * 8];   \
      b[j].q[1] = *(const uint4*)&Bs[bufi][row * LSTR + 16 + half * 8]; \
    }                                                                \
    _Pragma("unroll") for (int i = 0; i < 2; ++i)                    \
      _Pragma("unroll") for (int j = 0; j < 4; ++j)                  \
        acc[i][j] = wmma_bf16(a[i].v, b[j].v, acc[i][j]);            \
  } while (0)

  // Prologue: tile k=0 into buffer 0.
  GEMM_ISSUE(0, 0);

  for (int k0 = 0; k0 < K; k0 += 2 * GTK) {
    // ---- first half: stage tile k0+32 into buf1, compute buf0 ----
    GEMM_ISSUE(1, k0 + GTK);     // k0+32 < K always (K multiple of 64)
    WAIT_ASYNC(4);               // retire tile k0 (oldest 4 ops)
    __syncthreads();
    GEMM_COMPUTE(0);
    __syncthreads();             // buf0 free for refill

    // ---- second half: stage tile k0+64 into buf0, compute buf1 ----
    if (k0 + 2 * GTK < K) {
      GEMM_ISSUE(0, k0 + 2 * GTK);
      WAIT_ASYNC(4);             // retire tile k0+32
    } else {
      WAIT_ASYNC(0);
    }
    __syncthreads();
    GEMM_COMPUTE(1);
    __syncthreads();             // buf1 free for refill
  }

  // Epilogue. C layout: VGPR r holds row (r + half*8), col l16.
#pragma unroll
  for (int i = 0; i < 2; ++i) {
#pragma unroll
    for (int j = 0; j < 4; ++j) {
      int n = n0 + wn * 64 + j * 16 + l16;
      float bv = bias[n];
#pragma unroll
      for (int r = 0; r < 8; ++r) {
        int m = m0 + wm * 32 + i * 16 + r + half * 8;
        size_t o = (size_t)m * N + n;
        float v = acc[i][j][r] + bv;
        if (mode == 0) {
          ((__bf16*)outp)[o] = (__bf16)v;
        } else if (mode == 1) {
          ((float*)outp)[o] = v + residual[o];
        } else {
          float g = 0.5f * v * (1.0f + erff(v * 0.70710678118654752f));
          ((__bf16*)outp)[o] = (__bf16)g;
        }
      }
    }
  }
}

// ---------------------------------------------------------------------------
// Flash attention with lag-time score scaling + position bias.
// Grid (S/128, H, B); 8 waves/block; each wave owns 16 query rows and
// streams keys in chunks of 32. scores = (Q.K^T)/scale + pbias, online
// softmax, ctx += P.V.  scale = 9 - 1/(max(lag,0)+1), lag=(tq-tk)/60000.
// ---------------------------------------------------------------------------
__global__ __launch_bounds__(256) void attn_kernel(
    const __bf16* __restrict__ Q, const __bf16* __restrict__ Kb,
    const __bf16* __restrict__ V, const float* __restrict__ pbias,
    const long long* __restrict__ ts, __bf16* __restrict__ ctx) {
  const int qt = blockIdx.x, h = blockIdx.y, b = blockIdx.z;
  const int tid  = threadIdx.x;
  const int lane = tid & 31;
  const int wave = tid >> 5;
  const int half = lane >> 4;
  const int l16  = lane & 15;
  const int q0   = qt * 128 + wave * 16;          // wave's 16 query rows
  const size_t tokQ = (size_t)b * S_ + q0;

  __shared__ __bf16 Vt[DH_ * LSTR];               // V^T chunk: [d][key]
  __shared__ __bf16 Pl[8][16 * LSTR];             // per-wave P tile bf16

  // Q fragments for both K=32 steps of the d=64 dot product (held in regs)
  Frag aq[2];
#pragma unroll
  for (int d = 0; d < 2; ++d) {
    const __bf16* qp = Q + (tokQ + l16) * D_ + h * DH_ + d * 32;
    aq[d].q[0] = *(const uint4*)(qp + half * 8);
    aq[d].q[1] = *(const uint4*)(qp + 16 + half * 8);
  }
  float tq[8];
#pragma unroll
  for (int r = 0; r < 8; ++r)
    tq[r] = (float)ts[tokQ + r + half * 8];

  float mrow[8], lrow[8];
#pragma unroll
  for (int r = 0; r < 8; ++r) { mrow[r] = -3.0e38f; lrow[r] = 0.0f; }
  v8f acco[4];
#pragma unroll
  for (int j = 0; j < 4; ++j)
    acco[j] = (v8f){0.f, 0.f, 0.f, 0.f, 0.f, 0.f, 0.f, 0.f};

#pragma unroll 2
  for (int kk = 0; kk < S_; kk += 32) {
    // ---- stage V^T chunk (32 keys x 64 d) into LDS, transposed.
    // Batch the global b64 loads into registers first, then scatter the
    // 2-byte LDS stores, so the vmem waits are merged.
    __syncthreads();
    U2B vreg[2];
#pragma unroll
    for (int i = 0; i < 2; ++i) {
      int e   = tid + i * 256;        // 512 groups of 4 d-elements
      int d   = (e * 4) & 63;
      int key = (e * 4) >> 6;
      vreg[i].u = *(const uint2*)(V + ((size_t)b * S_ + kk + key) * D_ +
                                  h * DH_ + d);
    }
#pragma unroll
    for (int i = 0; i < 2; ++i) {
      int e   = tid + i * 256;
      int d   = (e * 4) & 63;
      int key = (e * 4) >> 6;
#pragma unroll
      for (int u = 0; u < 4; ++u) Vt[(d + u) * LSTR + key] = vreg[i].h[u];
    }
    __syncthreads();

    // ---- scores: 16q x 32k, accumulate over d in two K=32 WMMA steps ----
    v8f sc[2];
    sc[0] = (v8f){0.f, 0.f, 0.f, 0.f, 0.f, 0.f, 0.f, 0.f};
    sc[1] = sc[0];
#pragma unroll
    for (int d = 0; d < 2; ++d) {
#pragma unroll
      for (int nt = 0; nt < 2; ++nt) {
        Frag bk;  // B-frag straight from global K: lane=key, contiguous d
        const __bf16* kp =
            Kb + ((size_t)b * S_ + kk + nt * 16 + l16) * D_ + h * DH_ +
            d * 32 + half * 8;
        bk.q[0] = *(const uint4*)kp;
        bk.q[1] = *(const uint4*)(kp + 16);
        sc[nt] = wmma_bf16(aq[d].v, bk.v, sc[nt]);
      }
    }

    float tk[2];
#pragma unroll
    for (int nt = 0; nt < 2; ++nt)
      tk[nt] = (float)ts[(size_t)b * S_ + kk + nt * 16 + l16];

    // ---- per-row online softmax update ----
#pragma unroll
    for (int r = 0; r < 8; ++r) {
      int qi = q0 + r + half * 8;
      float vals[2];
#pragma unroll
      for (int nt = 0; nt < 2; ++nt) {
        float lag = (tq[r] - tk[nt]) * (1.0f / 60000.0f);
        float scale = 9.0f - 1.0f / (fmaxf(lag, 0.0f) + 1.0f);
        int ki = kk + nt * 16 + l16;
        float pb = pbias[((size_t)h * S_ + qi) * S_ + ki];
        vals[nt] = sc[nt][r] / scale + pb;
      }
      float cmax = fmaxf(vals[0], vals[1]);
#pragma unroll
      for (int off = 1; off < 16; off <<= 1)
        cmax = fmaxf(cmax, __shfl_xor(cmax, off, 16));
      float newm = fmaxf(mrow[r], cmax);
      float corr = __expf(mrow[r] - newm);
      float e0 = __expf(vals[0] - newm);
      float e1 = __expf(vals[1] - newm);
      float rs = e0 + e1;
#pragma unroll
      for (int off = 1; off < 16; off <<= 1)
        rs += __shfl_xor(rs, off, 16);
      lrow[r] = lrow[r] * corr + rs;
      mrow[r] = newm;
#pragma unroll
      for (int j = 0; j < 4; ++j) acco[j][r] *= corr;
      // park P (bf16) in wave-private LDS, C-layout -> A-layout bridge
      Pl[wave][(r + half * 8) * LSTR + l16]      = (__bf16)e0;
      Pl[wave][(r + half * 8) * LSTR + 16 + l16] = (__bf16)e1;
    }

    // ---- ctx += P(16x32) . Vchunk(32x64) ----
    Frag ap;
    ap.q[0] = *(const uint4*)&Pl[wave][l16 * LSTR + half * 8];
    ap.q[1] = *(const uint4*)&Pl[wave][l16 * LSTR + 16 + half * 8];
#pragma unroll
    for (int j = 0; j < 4; ++j) {
      Frag bv;
      bv.q[0] = *(const uint4*)&Vt[(j * 16 + l16) * LSTR + half * 8];
      bv.q[1] = *(const uint4*)&Vt[(j * 16 + l16) * LSTR + 16 + half * 8];
      acco[j] = wmma_bf16(ap.v, bv.v, acco[j]);
    }
  }

  // ---- normalize and write ctx (merged heads, bf16) ----
#pragma unroll
  for (int j = 0; j < 4; ++j)
#pragma unroll
    for (int r = 0; r < 8; ++r) {
      size_t o = ((size_t)b * S_ + q0 + r + half * 8) * D_ + h * DH_ +
                 j * 16 + l16;
      ctx[o] = (__bf16)(acco[j][r] / lrow[r]);
    }
}

// ---------------------------------------------------------------------------
// LayerNorm over D=512; one block per row; emits f32 (residual stream)
// and bf16 (next GEMM input).
// ---------------------------------------------------------------------------
__global__ __launch_bounds__(256) void ln_kernel(
    const float* __restrict__ x, const float* __restrict__ g,
    const float* __restrict__ bt, float* __restrict__ outf,
    __bf16* __restrict__ outb) {
  __shared__ float red[256];
  const int tid = threadIdx.x;
  const float* xr = x + (size_t)blockIdx.x * D_;
  float v0 = xr[tid], v1 = xr[tid + 256];

  red[tid] = v0 + v1;
  __syncthreads();
  for (int o = 128; o > 0; o >>= 1) {
    if (tid < o) red[tid] += red[tid + o];
    __syncthreads();
  }
  float mu = red[0] * (1.0f / D_);
  __syncthreads();

  float d0 = v0 - mu, d1 = v1 - mu;
  red[tid] = d0 * d0 + d1 * d1;
  __syncthreads();
  for (int o = 128; o > 0; o >>= 1) {
    if (tid < o) red[tid] += red[tid + o];
    __syncthreads();
  }
  float rs = rsqrtf(red[0] * (1.0f / D_) + 1e-12f);

  float y0 = d0 * rs * g[tid] + bt[tid];
  float y1 = d1 * rs * g[tid + 256] + bt[tid + 256];
  float* of = outf + (size_t)blockIdx.x * D_;
  __bf16* ob = outb + (size_t)blockIdx.x * D_;
  of[tid] = y0;       of[tid + 256] = y1;
  ob[tid] = (__bf16)y0; ob[tid + 256] = (__bf16)y1;
}

// ---------------------------------------------------------------------------
// Host-side orchestration (all launches on `stream`, no allocs/syncs)
// ---------------------------------------------------------------------------
extern "C" void kernel_launch(void* const* d_in, const int* in_sizes, int n_in,
                              void* d_out, int out_size, void* d_ws,
                              size_t ws_size, hipStream_t stream) {
  const float*     hidden = (const float*)d_in[0];
  const float*     pbias  = (const float*)d_in[1];
  const long long* tstamp = (const long long*)d_in[2];
  const float* Wq = (const float*)d_in[3];
  const float* bq = (const float*)d_in[4];
  const float* Wk = (const float*)d_in[5];
  const float* bk = (const float*)d_in[6];
  const float* Wv = (const float*)d_in[7];
  const float* bv = (const float*)d_in[8];
  const float* Wo = (const float*)d_in[9];
  const float* bo = (const float*)d_in[10];
  const float* g1 = (const float*)d_in[11];
  const float* b1 = (const float*)d_in[12];
  const float* Wi = (const float*)d_in[13];
  const float* bi = (const float*)d_in[14];
  const float* Wf = (const float*)d_in[15];
  const float* bf = (const float*)d_in[16];
  const float* g2 = (const float*)d_in[17];
  const float* b2 = (const float*)d_in[18];
  float* out = (float*)d_out;  // doubles as the f32 residual stream x

  const size_t MD  = (size_t)M_ * D_;
  const size_t DD2 = (size_t)D_ * D_ * 2;
  const size_t DF2 = (size_t)D_ * F_ * 2;

  char* ws = (char*)d_ws;
  __bf16* xb    = (__bf16*)(ws);                  // [M,D] bf16
  __bf16* qb    = (__bf16*)(ws + 1 * MD * 2);
  __bf16* kb    = (__bf16*)(ws + 2 * MD * 2);
  __bf16* vb    = (__bf16*)(ws + 3 * MD * 2);
  __bf16* cb    = (__bf16*)(ws + 4 * MD * 2);     // ctx
  __bf16* hb    = qb;                             // [M,F] reuses q..ctx span
  __bf16* ab    = (__bf16*)(ws + 5 * MD * 2);     // attn bf16
  float*  pre   = (float*)(ws + 6 * MD * 2);      // [M,D] f32 pre-LN
  float*  attnf = (float*)(ws + 6 * MD * 2 + MD * 4);
  char*   wbase = ws + 6 * MD * 2 + 2 * MD * 4;
  __bf16* wqt = (__bf16*)(wbase);
  __bf16* wkt = (__bf16*)(wbase + 1 * DD2);
  __bf16* wvt = (__bf16*)(wbase + 2 * DD2);
  __bf16* wot = (__bf16*)(wbase + 3 * DD2);
  __bf16* wit = (__bf16*)(wbase + 4 * DD2);
  __bf16* wft = (__bf16*)(wbase + 4 * DD2 + DF2);

  dim3 blk(256);
  dim3 gD(D_ / GTN, M_ / GTM);   // GEMMs producing [M,512]
  dim3 gF(F_ / GTN, M_ / GTM);   // GEMM producing [M,2048]
  dim3 gAttn(S_ / 128, H_, B_);
  int  gwDD = (int)(((size_t)D_ * D_ + 255) / 256);
  int  gwDF = (int)(((size_t)D_ * F_ + 255) / 256);

  cvt_f32_bf16_kernel<<<(int)(MD / 1024), blk, 0, stream>>>(hidden, xb, MD);

  for (int l = 0; l < L_; ++l) {
    const size_t oDD = (size_t)l * D_ * D_;
    const size_t oDF = (size_t)l * D_ * F_;
    wtrans_kernel<<<gwDD, blk, 0, stream>>>(Wq + oDD, wqt, D_, D_);
    wtrans_kernel<<<gwDD, blk, 0, stream>>>(Wk + oDD, wkt, D_, D_);
    wtrans_kernel<<<gwDD, blk, 0, stream>>>(Wv + oDD, wvt, D_, D_);
    wtrans_kernel<<<gwDD, blk, 0, stream>>>(Wo + oDD, wot, D_, D_);
    wtrans_kernel<<<gwDF, blk, 0, stream>>>(Wi + oDF, wit, D_, F_);
    wtrans_kernel<<<gwDF, blk, 0, stream>>>(Wf + oDF, wft, F_, D_);

    // Q,K,V projections (bias fused, bf16 out)
    gemm_bf16_kernel<<<gD, blk, 0, stream>>>(xb, wqt, bq + l * D_, nullptr,
                                             qb, M_, D_, D_, 0);
    gemm_bf16_kernel<<<gD, blk, 0, stream>>>(xb, wkt, bk + l * D_, nullptr,
                                             kb, M_, D_, D_, 0);
    gemm_bf16_kernel<<<gD, blk, 0, stream>>>(xb, wvt, bv + l * D_, nullptr,
                                             vb, M_, D_, D_, 0);

    attn_kernel<<<gAttn, blk, 0, stream>>>(qb, kb, vb, pbias, tstamp, cb);

    // O projection + residual -> pre-LN sum, then LN1
    const float* resid0 = (l == 0) ? hidden : out;
    gemm_bf16_kernel<<<gD, blk, 0, stream>>>(cb, wot, bo + l * D_, resid0,
                                             pre, M_, D_, D_, 1);
    ln_kernel<<<M_, blk, 0, stream>>>(pre, g1 + l * D_, b1 + l * D_, attnf, ab);

    // FFN: Wi (gelu fused) then Wf (+residual), then LN2 -> out(f32) + xb(bf16)
    gemm_bf16_kernel<<<gF, blk, 0, stream>>>(ab, wit, bi + l * F_, nullptr,
                                             hb, M_, F_, D_, 2);
    gemm_bf16_kernel<<<gD, blk, 0, stream>>>(hb, wft, bf + l * D_, attnf,
                                             pre, M_, D_, F_, 1);
    ln_kernel<<<M_, blk, 0, stream>>>(pre, g2 + l * D_, b2 + l * D_, out, xb);
  }
}
